// LSTMBaseline_87969520157108
// MI455X (gfx1250) — matrix operand order repaired and checked by reading, not compiled, not run on hardware
//
#include <hip/hip_runtime.h>

typedef __attribute__((ext_vector_type(16))) _Float16 v16h;
typedef __attribute__((ext_vector_type(8)))  _Float16 v8h;
typedef __attribute__((ext_vector_type(8)))  float    v8f;

#define WAVES   4
#define NNODES  862
#define NB      128
#define BN      (NB*NNODES)
#define T_ENC   24
#define HORIZON 12

// D = A(16x32 f16) * B(32x16 f16) + C(16x16 f32)
__device__ __forceinline__ v8f wmma16(v16h a, v16h b, v8f c) {
  return __builtin_amdgcn_wmma_f32_16x16x32_f16(false, a, false, b, (short)0, c, false, false);
}

// A-fragment layout (contiguous, matches B loads): lane<16 holds K={k0..k0+15},
// lane>=16 holds K={k0+16..k0+31}, row M = lane&15.
__device__ __forceinline__ v16h afrag(const _Float16* row, int k0, int kh) {
  const _Float16* p = row + k0 + (kh << 4);
  v8h lo = *(const v8h*)p;
  v8h hi = *(const v8h*)(p + 8);
  return __builtin_shufflevector(lo, hi, 0,1,2,3,4,5,6,7,8,9,10,11,12,13,14,15);
}

__device__ __forceinline__ float sig_f(float x)  { return 1.0f / (1.0f + __expf(-x)); }
__device__ __forceinline__ float tanh_f(float x) { float e = __expf(2.0f * x); return (e - 1.0f) / (e + 1.0f); }

// intra-wave LDS ordering: same-wave DS ops are in-order; wait + compiler fence
__device__ __forceinline__ void wave_sync() { asm volatile("s_wait_dscnt 0x0" ::: "memory"); }

__global__ __launch_bounds__(WAVES * 32)
void lstm_seq2seq_wmma(const float* __restrict__ traffic,
                       const float* __restrict__ eWih0, const float* __restrict__ eWhh0, const float* __restrict__ eb0,
                       const float* __restrict__ eWih1, const float* __restrict__ eWhh1, const float* __restrict__ eb1,
                       const float* __restrict__ dWih0, const float* __restrict__ dWhh0, const float* __restrict__ db0,
                       const float* __restrict__ dWih1, const float* __restrict__ dWhh1, const float* __restrict__ db1,
                       const float* __restrict__ headW, const float* __restrict__ headB,
                       float* __restrict__ out)
{
  // Phase-shared big weights (f16, row-major [gate_row 256][k 64]) -> 96 KB
  __shared__ _Float16 W_l0[256][64];   // enc_Whh0 -> dec_Whh0
  __shared__ _Float16 W_i1[256][64];   // enc_Wih1 -> dec_Wih1
  __shared__ _Float16 W_h1[256][64];   // enc_Whh1 -> dec_Whh1
  __shared__ float w0s[256 * 3];       // enc_Wih0 (K=3 handled post-WMMA)
  __shared__ float b0s[256];
  __shared__ float b1s[256];
  __shared__ float dw0s[256];          // dec_Wih0 (K=1)
  __shared__ float hWs[64];
  __shared__ float hb_s;
  // Per-wave recurrent state: h (f16, feeds WMMA A) and c (f32, lane-private)
  __shared__ _Float16 hbuf0[WAVES][16][64];
  __shared__ _Float16 hbuf1[WAVES][16][64];
  __shared__ float    cbuf0[WAVES][16][64];
  __shared__ float    cbuf1[WAVES][16][64];
  __shared__ float    xsb[WAVES][16][4]; // x features [0..2], autoregressive input [3]

  const int tid  = threadIdx.x;
  const int wv   = tid >> 5;
  const int lane = tid & 31;
  const int kh   = lane >> 4;   // lane half (0/1)
  const int lp   = lane & 15;

  // ---- stage encoder weights (cooperative, f32 -> f16) + zero state ----
  for (int i = tid; i < 256 * 64; i += WAVES * 32) {
    (&W_l0[0][0])[i] = (_Float16)eWhh0[i];
    (&W_i1[0][0])[i] = (_Float16)eWih1[i];
    (&W_h1[0][0])[i] = (_Float16)eWhh1[i];
  }
  for (int i = tid; i < 768; i += WAVES * 32) w0s[i] = eWih0[i];
  for (int i = tid; i < 256; i += WAVES * 32) { b0s[i] = eb0[i]; b1s[i] = eb1[i]; }
  for (int i = tid; i < WAVES * 16 * 64; i += WAVES * 32) {
    (&hbuf0[0][0][0])[i] = (_Float16)0.0f;
    (&hbuf1[0][0][0])[i] = (_Float16)0.0f;
    (&cbuf0[0][0][0])[i] = 0.0f;
    (&cbuf1[0][0][0])[i] = 0.0f;
  }
  __syncthreads();

  _Float16* h0w = &hbuf0[wv][0][0];
  _Float16* h1w = &hbuf1[wv][0][0];
  float*    c0w = &cbuf0[wv][0][0];
  float*    c1w = &cbuf1[wv][0][0];
  float*    xsw = &xsb[wv][0][0];

  const int sIdx = (blockIdx.x * WAVES + wv) * 16 + lp;  // flat sequence id (b*N + node)
  const int sB = sIdx / NNODES;
  const int sN = sIdx - sB * NNODES;

  const v8f zf = {};   // zero accumulator -> SRC2 inline 0 on first WMMA of chain

  // =========================== ENCODER ===========================
  for (int t = 0; t < T_ENC; ++t) {
    if (kh == 0) {
      const float* p = traffic + (((size_t)sB * T_ENC + t) * NNODES + sN) * 3;
      xsw[lp * 4 + 0] = p[0];
      xsw[lp * 4 + 1] = p[1];
      xsw[lp * 4 + 2] = p[2];
      if (t + 1 < T_ENC)
        __builtin_prefetch(traffic + (((size_t)sB * T_ENC + t + 1) * NNODES + sN) * 3, 0, 1);
    }
    wave_sync();

    float xr0[8], xr1[8], xr2[8];
    #pragma unroll
    for (int r = 0; r < 8; ++r) {
      const int m = kh * 8 + r;
      xr0[r] = xsw[m * 4 + 0]; xr1[r] = xsw[m * 4 + 1]; xr2[r] = xsw[m * 4 + 2];
    }

    { // ---- encoder layer 0: WMMA h0@Whh0^T, then bias + x@Wih0^T post-accumulate ----
      const _Float16* h0a = h0w + lp * 64;
      v16h a0 = afrag(h0a, 0, kh);
      v16h a1 = afrag(h0a, 32, kh);
      #pragma unroll
      for (int g = 0; g < 4; ++g) {
        v8f acc[4];
        float bq[4], wf0[4], wf1[4], wf2[4];
        #pragma unroll
        for (int q = 0; q < 4; ++q) {              // q: gate i,f,g,o ; tile = 4q+g
          const int n = 16 * (4 * q + g) + lp;
          const _Float16* Wn = &W_l0[n][0];
          bq[q] = b0s[n]; wf0[q] = w0s[n * 3]; wf1[q] = w0s[n * 3 + 1]; wf2[q] = w0s[n * 3 + 2];
          v8f c = wmma16(a0, *(const v16h*)(Wn + kh * 16), zf);
          acc[q] = wmma16(a1, *(const v16h*)(Wn + 32 + kh * 16), c);
        }
        #pragma unroll
        for (int r = 0; r < 8; ++r) {              // elementwise LSTM in D layout
          const int off = (kh * 8 + r) * 64 + 16 * g + lp;
          const float iv = acc[0][r] + bq[0] + xr0[r] * wf0[0] + xr1[r] * wf1[0] + xr2[r] * wf2[0];
          const float fv = acc[1][r] + bq[1] + xr0[r] * wf0[1] + xr1[r] * wf1[1] + xr2[r] * wf2[1];
          const float gv = acc[2][r] + bq[2] + xr0[r] * wf0[2] + xr1[r] * wf1[2] + xr2[r] * wf2[2];
          const float ov = acc[3][r] + bq[3] + xr0[r] * wf0[3] + xr1[r] * wf1[3] + xr2[r] * wf2[3];
          const float cn = sig_f(fv) * c0w[off] + sig_f(iv) * tanh_f(gv);
          c0w[off] = cn;
          h0w[off] = (_Float16)(sig_f(ov) * tanh_f(cn));
        }
      }
    }
    wave_sync();

    { // ---- encoder layer 1: h0new@Wih1^T + h1@Whh1^T (4 WMMAs/tile), bias post ----
      const _Float16* h0a = h0w + lp * 64;
      const _Float16* h1a = h1w + lp * 64;
      v16h ai0 = afrag(h0a, 0, kh), ai1 = afrag(h0a, 32, kh);
      v16h ah0 = afrag(h1a, 0, kh), ah1 = afrag(h1a, 32, kh);
      #pragma unroll
      for (int g = 0; g < 4; ++g) {
        v8f acc[4];
        float bq[4];
        #pragma unroll
        for (int q = 0; q < 4; ++q) {
          const int n = 16 * (4 * q + g) + lp;
          bq[q] = b1s[n];
          v8f c = wmma16(ai0, *(const v16h*)(&W_i1[n][0] + kh * 16), zf);
          c = wmma16(ai1, *(const v16h*)(&W_i1[n][0] + 32 + kh * 16), c);
          c = wmma16(ah0, *(const v16h*)(&W_h1[n][0] + kh * 16), c);
          acc[q] = wmma16(ah1, *(const v16h*)(&W_h1[n][0] + 32 + kh * 16), c);
        }
        #pragma unroll
        for (int r = 0; r < 8; ++r) {
          const int off = (kh * 8 + r) * 64 + 16 * g + lp;
          const float iv = acc[0][r] + bq[0];
          const float fv = acc[1][r] + bq[1];
          const float gv = acc[2][r] + bq[2];
          const float ov = acc[3][r] + bq[3];
          const float cn = sig_f(fv) * c1w[off] + sig_f(iv) * tanh_f(gv);
          c1w[off] = cn;
          h1w[off] = (_Float16)(sig_f(ov) * tanh_f(cn));
        }
      }
    }
    wave_sync();
  }

  // decoder first input = last observed speed (t=23, feature 0), still in xsw
  if (kh == 0) xsw[lp * 4 + 3] = xsw[lp * 4 + 0];

  // ---- phase switch: overwrite the 96 KB weight region with decoder weights ----
  __syncthreads();
  for (int i = tid; i < 256 * 64; i += WAVES * 32) {
    (&W_l0[0][0])[i] = (_Float16)dWhh0[i];
    (&W_i1[0][0])[i] = (_Float16)dWih1[i];
    (&W_h1[0][0])[i] = (_Float16)dWhh1[i];
  }
  for (int i = tid; i < 256; i += WAVES * 32) { b0s[i] = db0[i]; b1s[i] = db1[i]; dw0s[i] = dWih0[i]; }
  for (int i = tid; i < 64; i += WAVES * 32) hWs[i] = headW[i];
  if (tid == 0) hb_s = headB[0];
  __syncthreads();

  // =========================== DECODER ===========================
  for (int hh = 0; hh < HORIZON; ++hh) {
    float xin[8];
    #pragma unroll
    for (int r = 0; r < 8; ++r) xin[r] = xsw[(kh * 8 + r) * 4 + 3];

    { // ---- decoder layer 0 (scalar input K=1 added post-WMMA) ----
      const _Float16* h0a = h0w + lp * 64;
      v16h a0 = afrag(h0a, 0, kh);
      v16h a1 = afrag(h0a, 32, kh);
      #pragma unroll
      for (int g = 0; g < 4; ++g) {
        v8f acc[4];
        float bq[4], wi[4];
        #pragma unroll
        for (int q = 0; q < 4; ++q) {
          const int n = 16 * (4 * q + g) + lp;
          const _Float16* Wn = &W_l0[n][0];
          bq[q] = b0s[n]; wi[q] = dw0s[n];
          v8f c = wmma16(a0, *(const v16h*)(Wn + kh * 16), zf);
          acc[q] = wmma16(a1, *(const v16h*)(Wn + 32 + kh * 16), c);
        }
        #pragma unroll
        for (int r = 0; r < 8; ++r) {
          const int off = (kh * 8 + r) * 64 + 16 * g + lp;
          const float iv = acc[0][r] + bq[0] + xin[r] * wi[0];
          const float fv = acc[1][r] + bq[1] + xin[r] * wi[1];
          const float gv = acc[2][r] + bq[2] + xin[r] * wi[2];
          const float ov = acc[3][r] + bq[3] + xin[r] * wi[3];
          const float cn = sig_f(fv) * c0w[off] + sig_f(iv) * tanh_f(gv);
          c0w[off] = cn;
          h0w[off] = (_Float16)(sig_f(ov) * tanh_f(cn));
        }
      }
    }
    wave_sync();

    { // ---- decoder layer 1 ----
      const _Float16* h0a = h0w + lp * 64;
      const _Float16* h1a = h1w + lp * 64;
      v16h ai0 = afrag(h0a, 0, kh), ai1 = afrag(h0a, 32, kh);
      v16h ah0 = afrag(h1a, 0, kh), ah1 = afrag(h1a, 32, kh);
      #pragma unroll
      for (int g = 0; g < 4; ++g) {
        v8f acc[4];
        float bq[4];
        #pragma unroll
        for (int q = 0; q < 4; ++q) {
          const int n = 16 * (4 * q + g) + lp;
          bq[q] = b1s[n];
          v8f c = wmma16(ai0, *(const v16h*)(&W_i1[n][0] + kh * 16), zf);
          c = wmma16(ai1, *(const v16h*)(&W_i1[n][0] + 32 + kh * 16), c);
          c = wmma16(ah0, *(const v16h*)(&W_h1[n][0] + kh * 16), c);
          acc[q] = wmma16(ah1, *(const v16h*)(&W_h1[n][0] + 32 + kh * 16), c);
        }
        #pragma unroll
        for (int r = 0; r < 8; ++r) {
          const int off = (kh * 8 + r) * 64 + 16 * g + lp;
          const float iv = acc[0][r] + bq[0];
          const float fv = acc[1][r] + bq[1];
          const float gv = acc[2][r] + bq[2];
          const float ov = acc[3][r] + bq[3];
          const float cn = sig_f(fv) * c1w[off] + sig_f(iv) * tanh_f(gv);
          c1w[off] = cn;
          h1w[off] = (_Float16)(sig_f(ov) * tanh_f(cn));
        }
      }
    }
    wave_sync();

    // ---- head: pred = h1 @ headW^T + b ; write output + feed back ----
    if (kh == 0) {
      float a = hb_s;
      #pragma unroll
      for (int j = 0; j < 64; ++j) a += (float)h1w[lp * 64 + j] * hWs[j];
      out[((size_t)sB * HORIZON + hh) * NNODES + sN] = a;
      xsw[lp * 4 + 3] = a;
    }
    wave_sync();
  }
}

extern "C" void kernel_launch(void* const* d_in, const int* in_sizes, int n_in,
                              void* d_out, int out_size, void* d_ws, size_t ws_size,
                              hipStream_t stream) {
  (void)in_sizes; (void)n_in; (void)out_size; (void)d_ws; (void)ws_size;
  const float* traffic = (const float*)d_in[0];
  const float* eWih0   = (const float*)d_in[1];
  const float* eWhh0   = (const float*)d_in[2];
  const float* eb0     = (const float*)d_in[3];
  const float* eWih1   = (const float*)d_in[4];
  const float* eWhh1   = (const float*)d_in[5];
  const float* eb1     = (const float*)d_in[6];
  const float* dWih0   = (const float*)d_in[7];
  const float* dWhh0   = (const float*)d_in[8];
  const float* db0     = (const float*)d_in[9];
  const float* dWih1   = (const float*)d_in[10];
  const float* dWhh1   = (const float*)d_in[11];
  const float* db1     = (const float*)d_in[12];
  const float* headW   = (const float*)d_in[13];
  const float* headB   = (const float*)d_in[14];
  float* out = (float*)d_out;

  const int nblocks = BN / (16 * WAVES);  // 110336 / 64 = 1724
  lstm_seq2seq_wmma<<<nblocks, WAVES * 32, 0, stream>>>(
      traffic, eWih0, eWhh0, eb0, eWih1, eWhh1, eb1,
      dWih0, dWhh0, db0, dWih1, dWhh1, db1, headW, headB, out);
}
